// FixedGeneAttentionLayer_21345987461806
// MI455X (gfx1250) — matrix operand order repaired and checked by reading, not compiled, not run on hardware
//
#include <hip/hip_runtime.h>
#include <hip/hip_bf16.h>

typedef _Float16 h16;
typedef __attribute__((ext_vector_type(16))) _Float16 v16h;
typedef __attribute__((ext_vector_type(8)))  _Float16 v8h;
typedef __attribute__((ext_vector_type(4)))  _Float16 v4h;
typedef __attribute__((ext_vector_type(8)))  float    v8f;
typedef __attribute__((ext_vector_type(4)))  float    v4f;

#define WMMA_F16(a, b, c) \
  __builtin_amdgcn_wmma_f32_16x16x32_f16(false, (a), false, (b), (short)0, (c), false, false)

union AFrag { v16h v; v8h h[2]; };

// ---- WMMA fragment loaders -------------------------------------------------
// A matrix 16x32 f16 (row-major source, stride lda elems, ptr at [row0,k0]):
// lanes 0-15: M=lane, K [0..15]; lanes 16-31: M=lane-16, K [16..31]
__device__ inline v16h load_a_g(const h16* __restrict__ p, int lda, int lane) {
  int row = lane & 15;
  int kb  = (lane >> 4) << 4;
  const h16* q = p + row * lda + kb;
  AFrag f;
  f.h[0] = *(const v8h*)(q);
  f.h[1] = *(const v8h*)(q + 8);
  return f.v;
}

// B matrix 32x16 f16 (NT gemm: conceptual B[k][n], column n = contiguous source
// row n). Per lane: col = lane&15, 16 contiguous K halves at kb=(lane>>4)*16.
__device__ inline v16h load_b_g(const h16* __restrict__ p, int ldb, int lane) {
  int col = lane & 15;
  int kb  = (lane >> 4) << 4;
  return *(const v16h*)(p + col * ldb + kb);
}

// ---- kernel 1: f32 -> f16 convert, 4-wide ---------------------------------
__global__ void k_cvt4(const float* __restrict__ src, h16* __restrict__ dst, int n4) {
  int i = blockIdx.x * blockDim.x + threadIdx.x;
  if (i >= n4) return;
  v4f v = *(const v4f*)(src + (size_t)i * 4);
  v4h o;
  o[0] = (h16)v[0]; o[1] = (h16)v[1]; o[2] = (h16)v[2]; o[3] = (h16)v[3];
  *(v4h*)(dst + (size_t)i * 4) = o;
}

// ---- kernel 1b: lm = log2(M) as f16, 4-wide (M in [0.05,1]: no denorms) ----
__global__ void k_log2m(const float* __restrict__ M, h16* __restrict__ lm, int n4) {
  int i = blockIdx.x * blockDim.x + threadIdx.x;
  if (i >= n4) return;
  v4f v = *(const v4f*)(M + (size_t)i * 4);
  v4h o;
#pragma unroll
  for (int j = 0; j < 4; ++j) o[j] = (h16)__builtin_amdgcn_logf(v[j]);
  *(v4h*)(lm + (size_t)i * 4) = o;
}

// ---- kernel 2: projection  y = (xh @ W^T) * mult, 16x64 per wave -----------
// X: [4096,1024] f16, W: [1024,1024] f16 row-major.
// MODE 0: out[(bh*2048 + s)*64 + d]   (Q, K layout [B,H,S,hd])
// MODE 1: out[(bh*64 + d)*2048 + s]   (V transposed [B,H,hd,S])
template <int MODE>
__global__ __launch_bounds__(256) void k_proj(const h16* __restrict__ X,
                                              const h16* __restrict__ W,
                                              h16* __restrict__ out, float mult) {
  const int K = 1024;
  int wave = threadIdx.x >> 5, lane = threadIdx.x & 31;
  int job = blockIdx.x * 8 + wave;          // 4096 jobs: 256 m-tiles x 16 n-groups
  int ng = job & 15, tm = job >> 4;
  int m0 = tm * 16, n0 = ng * 64;
  const h16* a_base = X + (size_t)m0 * K;
  const h16* b_base = W + (size_t)n0 * K;
  v8f acc0 = {}, acc1 = {}, acc2 = {}, acc3 = {};
#pragma unroll 2
  for (int kc = 0; kc < K; kc += 32) {
    v16h a  = load_a_g(a_base + kc, K, lane);
    v16h b0 = load_b_g(b_base + kc, K, lane);
    v16h b1 = load_b_g(b_base + (size_t)16 * K + kc, K, lane);
    v16h b2 = load_b_g(b_base + (size_t)32 * K + kc, K, lane);
    v16h b3 = load_b_g(b_base + (size_t)48 * K + kc, K, lane);
    acc0 = WMMA_F16(a, b0, acc0);
    acc1 = WMMA_F16(a, b1, acc1);
    acc2 = WMMA_F16(a, b2, acc2);
    acc3 = WMMA_F16(a, b3, acc3);
  }
  int rb = (lane >> 4) * 8;
  v8f accs[4] = {acc0, acc1, acc2, acc3};
#pragma unroll
  for (int j = 0; j < 4; ++j) {
    int coln = n0 + j * 16 + (lane & 15);
    int hh = coln >> 6, dd = coln & 63;
#pragma unroll
    for (int r = 0; r < 8; ++r) {
      int m = m0 + rb + r;
      int b = m >> 11, s = m & 2047;
      int bh = b * 16 + hh;
      h16 val = (h16)(accs[j][r] * mult);
      if (MODE == 0) out[((size_t)bh * 2048 + s) * 64 + dd] = val;
      else           out[((size_t)bh * 64 + dd) * 2048 + s] = val;
    }
  }
}

// ---- kernel 3: attention core ---------------------------------------------
// One workgroup (8 waves) per (b, h, 16-row q-tile); h fastest for M reuse.
// Q is pre-scaled by hd^-0.5 * log2(e), so phase 1 stores raw WMMA output and
// the whole softmax runs in base-2 (exact softmax invariance).
__global__ __launch_bounds__(256) void k_attn(const h16* __restrict__ Q,
                                              const h16* __restrict__ Km,
                                              const h16* __restrict__ Vt,
                                              const h16* __restrict__ LM,
                                              h16* __restrict__ AO) {
  const int S = 2048, HD = 64;
  __shared__ float sc[16][2048];        // 128 KB base-2 score strip
  __shared__ float s_inv[16];
  __shared__ float s_part[4][256];      // cross-wave reduce scratch (4 KB)

  int wave = threadIdx.x >> 5, lane = threadIdx.x & 31;
  int blk = blockIdx.x;                 // b*2048 + qt*16 + h
  int h  = blk & 15;
  int qt = (blk >> 4) & 127;
  int b  = blk >> 11;
  int bh = b * 16 + h;
  int q0 = qt * 16;

  // ---- phase 1: t = Q' K^T;  wave w -> keys [w*256, +256)
  const h16* qbase = Q + ((size_t)bh * S + q0) * HD;
  v16h a0 = load_a_g(qbase, HD, lane);
  v16h a1 = load_a_g(qbase + 32, HD, lane);
  int k0w = wave * 256;
  int rb = (lane >> 4) * 8;
  for (int kt = 0; kt < 256; kt += 16) {
    int kk = k0w + kt;
    const h16* kbase = Km + ((size_t)bh * S + kk) * HD;
    v16h b0 = load_b_g(kbase, HD, lane);
    v16h b1 = load_b_g(kbase + 32, HD, lane);
    v8f c = {};
    c = WMMA_F16(a0, b0, c);
    c = WMMA_F16(a1, b1, c);
    int key = kk + (lane & 15);
#pragma unroll
    for (int r = 0; r < 8; ++r) sc[rb + r][key] = c[r];
  }
  __syncthreads();

  // ---- phase 2: add log2-mask (coalesced v4h loads) + base-2 row softmax.
  // Wave w owns rows 2w, 2w+1; probs left unnormalized in LDS, 1/sum saved.
  for (int rr = 0; rr < 2; ++rr) {
    int row = wave * 2 + rr;
    const h16* lmrow = LM + ((size_t)b * S + q0 + row) * S;
    float mx = -3.0e38f;
    for (int k4 = lane * 4; k4 < S; k4 += 128) {
      v4f v = *(const v4f*)&sc[row][k4];
      v4h l = *(const v4h*)(lmrow + k4);
      v[0] += (float)l[0]; v[1] += (float)l[1];
      v[2] += (float)l[2]; v[3] += (float)l[3];
      *(v4f*)&sc[row][k4] = v;
      mx = fmaxf(mx, fmaxf(fmaxf(v[0], v[1]), fmaxf(v[2], v[3])));
    }
#pragma unroll
    for (int off = 16; off; off >>= 1) mx = fmaxf(mx, __shfl_xor(mx, off, 32));
    float sum = 0.f;
    for (int k4 = lane * 4; k4 < S; k4 += 128) {
      v4f v = *(const v4f*)&sc[row][k4];
      v4f p;
#pragma unroll
      for (int j = 0; j < 4; ++j) p[j] = __builtin_amdgcn_exp2f(v[j] - mx);
      *(v4f*)&sc[row][k4] = p;
      sum += p[0] + p[1] + p[2] + p[3];
    }
#pragma unroll
    for (int off = 16; off; off >>= 1) sum += __shfl_xor(sum, off, 32);
    if (lane == 0) s_inv[row] = 1.0f / sum;
  }
  __syncthreads();

  // ---- phase 3: out = attn @ V.  Wave -> (n-tile = wave&3, k-half = wave>>2)
  int nt = wave & 3;
  int khalf = wave >> 2;
  int arow = lane & 15;
  int akb = (lane >> 4) << 4;
  int kb16 = (lane >> 4) << 4;
  float inv = s_inv[arow];
  const h16* vrow = Vt + ((size_t)bh * HD + nt * 16 + (lane & 15)) * S;
  v8f acc = {};
  for (int kc = 0; kc < 1024; kc += 32) {
    int kg = khalf * 1024 + kc;
    v8f c0 = *(const v8f*)&sc[arow][kg + akb];
    v8f c1 = *(const v8f*)&sc[arow][kg + akb + 8];
    AFrag af;
#pragma unroll
    for (int i = 0; i < 8; ++i) { af.v[i] = (h16)(c0[i] * inv); af.v[8 + i] = (h16)(c1[i] * inv); }
    v16h bf = *(const v16h*)(vrow + kg + kb16);
    acc = WMMA_F16(af.v, bf, acc);
  }
  if (khalf == 1) {
#pragma unroll
    for (int r = 0; r < 8; ++r) s_part[nt][r * 32 + lane] = acc[r];
  }
  __syncthreads();
  if (khalf == 0) {
#pragma unroll
    for (int r = 0; r < 8; ++r) acc[r] += s_part[nt][r * 32 + lane];
    int coln = h * 64 + nt * 16 + (lane & 15);
#pragma unroll
    for (int r = 0; r < 8; ++r) {
      int s = q0 + rb + r;
      AO[((size_t)(b * S + s)) * 1024 + coln] = (h16)acc[r];
    }
  }
}

// ---- kernel 4: output projection  Y = AO @ Wo^T + bo, 16x64 per wave -------
__global__ __launch_bounds__(256) void k_out(const h16* __restrict__ AO,
                                             const h16* __restrict__ Wo,
                                             const float* __restrict__ bo,
                                             float* __restrict__ Y) {
  const int K = 1024;
  int wave = threadIdx.x >> 5, lane = threadIdx.x & 31;
  int job = blockIdx.x * 8 + wave;
  int ng = job & 15, tm = job >> 4;
  int m0 = tm * 16, n0 = ng * 64;
  const h16* a_base = AO + (size_t)m0 * K;
  const h16* b_base = Wo + (size_t)n0 * K;
  v8f acc0 = {}, acc1 = {}, acc2 = {}, acc3 = {};
#pragma unroll 2
  for (int kc = 0; kc < K; kc += 32) {
    v16h a  = load_a_g(a_base + kc, K, lane);
    v16h b0 = load_b_g(b_base + kc, K, lane);
    v16h b1 = load_b_g(b_base + (size_t)16 * K + kc, K, lane);
    v16h b2 = load_b_g(b_base + (size_t)32 * K + kc, K, lane);
    v16h b3 = load_b_g(b_base + (size_t)48 * K + kc, K, lane);
    acc0 = WMMA_F16(a, b0, acc0);
    acc1 = WMMA_F16(a, b1, acc1);
    acc2 = WMMA_F16(a, b2, acc2);
    acc3 = WMMA_F16(a, b3, acc3);
  }
  int rb = (lane >> 4) * 8;
  v8f accs[4] = {acc0, acc1, acc2, acc3};
#pragma unroll
  for (int j = 0; j < 4; ++j) {
    int coln = n0 + j * 16 + (lane & 15);
    float bias = bo[coln];
#pragma unroll
    for (int r = 0; r < 8; ++r) {
      int m = m0 + rb + r;
      Y[(size_t)m * 1024 + coln] = accs[j][r] + bias;
    }
  }
}

// ---- host side -------------------------------------------------------------
extern "C" void kernel_launch(void* const* d_in, const int* in_sizes, int n_in,
                              void* d_out, int out_size, void* d_ws, size_t ws_size,
                              hipStream_t stream) {
  (void)in_sizes; (void)n_in; (void)out_size; (void)ws_size;
  const float* x  = (const float*)d_in[0];   // [2,2048,1024]
  const float* M  = (const float*)d_in[1];   // [2,2048,2048]
  const float* Wq = (const float*)d_in[2];
  const float* Wk = (const float*)d_in[3];
  const float* Wv = (const float*)d_in[4];
  const float* Wo = (const float*)d_in[5];
  const float* bo = (const float*)d_in[6];
  float* Y = (float*)d_out;

  char* ws = (char*)d_ws;
  const size_t MB = 1024ull * 1024;
  h16* xh  = (h16*)(ws);               // 8 MB
  h16* wqh = (h16*)(ws + 8  * MB);     // 2 MB each
  h16* wkh = (h16*)(ws + 10 * MB);
  h16* wvh = (h16*)(ws + 12 * MB);
  h16* woh = (h16*)(ws + 14 * MB);
  h16* Qh  = (h16*)(ws + 16 * MB);     // 8 MB each
  h16* Kh  = (h16*)(ws + 24 * MB);
  h16* Vt  = (h16*)(ws + 32 * MB);
  h16* AO  = (h16*)(ws + 40 * MB);
  h16* LM  = (h16*)(ws + 48 * MB);     // 16 MB: log2(M) f16

  const int XN4 = (4096 * 1024) / 4, WN4 = (1024 * 1024) / 4, MN4 = (2 * 2048 * 2048) / 4;
  k_cvt4<<<XN4 / 256, 256, 0, stream>>>(x,  xh,  XN4);
  k_cvt4<<<WN4 / 256, 256, 0, stream>>>(Wq, wqh, WN4);
  k_cvt4<<<WN4 / 256, 256, 0, stream>>>(Wk, wkh, WN4);
  k_cvt4<<<WN4 / 256, 256, 0, stream>>>(Wv, wvh, WN4);
  k_cvt4<<<WN4 / 256, 256, 0, stream>>>(Wo, woh, WN4);
  k_log2m<<<MN4 / 256, 256, 0, stream>>>(M, LM, MN4);

  const float scale2 = 0.125f * 1.44269504088896f;  // hd^-0.5 * log2(e), folded into Q
  k_proj<0><<<512, 256, 0, stream>>>(xh, wqh, Qh, scale2);
  k_proj<0><<<512, 256, 0, stream>>>(xh, wkh, Kh, 1.0f);
  k_proj<1><<<512, 256, 0, stream>>>(xh, wvh, Vt, 1.0f);

  // 2*128*16 = 4096 (b, q-tile, h) workgroups, h fastest
  k_attn<<<4096, 256, 0, stream>>>(Qh, Kh, Vt, LM, AO);

  k_out<<<512, 256, 0, stream>>>(AO, woh, bo, Y);
}
